// TransModel_79104707658570
// MI455X (gfx1250) — matrix operand order repaired and checked by reading, not compiled, not run on hardware
//
#include <hip/hip_runtime.h>
#include <hip/hip_bf16.h>

typedef _Float16 half_t;
typedef __attribute__((ext_vector_type(16))) _Float16 v16h;
typedef __attribute__((ext_vector_type(8)))  _Float16 v8h;
typedef __attribute__((ext_vector_type(8)))  float    v8f;

#define P_TOT 11264   // 88 * 128 positions
#define SEQ   88
#define TT    128
#define CDIM  256

// ---------- WMMA fragment loaders (CDNA5 16x16x32 f16 layouts) ----------
// A 16x32: lanes hold M=lane&15, K = (lane>>4)*16 + [0..15] contiguous
__device__ __forceinline__ v16h ld_frag_a(const half_t* p) {
  v8h lo = *(const v8h*)p;
  v8h hi = *(const v8h*)(p + 8);
  return __builtin_shufflevector(lo, hi, 0,1,2,3,4,5,6,7,8,9,10,11,12,13,14,15);
}
// B 32x16 from Bt (N x K row-major): lane holds col=lane&15, K = (lane>>4)*16 + [0..15] contiguous
__device__ __forceinline__ v16h ld_frag_b(const half_t* p) {
  v8h lo = *(const v8h*)p;
  v8h hi = *(const v8h*)(p + 8);
  return __builtin_shufflevector(lo, hi, 0,1,2,3,4,5,6,7,8,9,10,11,12,13,14,15);
}

// ---------- generic f16 WMMA GEMM: C(MxN) = A(MxK) @ Bt(NxK)^T + bias (+res) ----------
// software-pipelined: fragments for k+32 load while the 4 WMMAs for k execute
__global__ __launch_bounds__(256)
void gemm_wmma(const half_t* __restrict__ A, const half_t* __restrict__ Bt,
               const float* __restrict__ bias, const float* __restrict__ res,
               float* __restrict__ C, half_t* __restrict__ Ch,
               int M, int N, int K) {
  int wid  = (blockIdx.x * blockDim.x + threadIdx.x) >> 5;
  int lane = threadIdx.x & 31;
  int n64  = N >> 6;
  int total = (M >> 4) * n64;
  if (wid >= total) return;              // wave-uniform
  int m0 = (wid / n64) << 4;
  int n0 = (wid % n64) << 6;

  const half_t* Arow = A + (size_t)(m0 + (lane & 15)) * K + ((lane >> 4) << 4);
  int bkb  = (lane >> 4) << 4;
  int bcol = lane & 15;
  const half_t* Bp[4];
#pragma unroll
  for (int nt = 0; nt < 4; nt++)
    Bp[nt] = Bt + (size_t)(n0 + nt * 16 + bcol) * K + bkb;

  v8f acc[4] = {};
  v16h a_cur = ld_frag_a(Arow);
  v16h b_cur[4];
#pragma unroll
  for (int nt = 0; nt < 4; nt++) b_cur[nt] = ld_frag_b(Bp[nt]);

  for (int k0 = 32; k0 < K; k0 += 32) {
    v16h a_nxt = ld_frag_a(Arow + k0);
    v16h b_nxt[4];
#pragma unroll
    for (int nt = 0; nt < 4; nt++) b_nxt[nt] = ld_frag_b(Bp[nt] + k0);
#pragma unroll
    for (int nt = 0; nt < 4; nt++)
      acc[nt] = __builtin_amdgcn_wmma_f32_16x16x32_f16(false, a_cur, false, b_cur[nt],
                                                       (short)0, acc[nt], false, false);
    a_cur = a_nxt;
#pragma unroll
    for (int nt = 0; nt < 4; nt++) b_cur[nt] = b_nxt[nt];
  }
#pragma unroll
  for (int nt = 0; nt < 4; nt++)
    acc[nt] = __builtin_amdgcn_wmma_f32_16x16x32_f16(false, a_cur, false, b_cur[nt],
                                                     (short)0, acc[nt], false, false);

  int rb = m0 + ((lane >> 4) << 3);
  int cc = lane & 15;
#pragma unroll
  for (int nt = 0; nt < 4; nt++) {
    int n = n0 + nt * 16 + cc;
    float bv = bias ? bias[n] : 0.f;
#pragma unroll
    for (int v = 0; v < 8; v++) {
      size_t idx = (size_t)(rb + v) * (size_t)N + n;
      float val = acc[nt][v] + bv;
      if (res) val += res[idx];
      C[idx] = val;
      if (Ch) Ch[idx] = (half_t)val;
    }
  }
}

// ---------- LSTM recurrent scan: one block per direction, h in LDS, c in regs ----------
__global__ __launch_bounds__(1024)
void lstm_scan(const float* __restrict__ G,      // (P_TOT, 1024): rows s*128+t, cols d*512 + gate*128 + jj
               const half_t* __restrict__ Whh,   // (2, 512, 128) f16, Bt layout
               float* __restrict__ Y) {          // (P_TOT, 256): col d*128+jj
  int d = blockIdx.x;
  int tid = threadIdx.x, wv = tid >> 5, lane = tid & 31;
  __shared__ half_t hbuf[2][96 * 128];           // 48 KB ping-pong
  for (int i = tid; i < 96 * 128; i += 1024) { hbuf[0][i] = (half_t)0.f; hbuf[1][i] = (half_t)0.f; }
  float creg[2][8];
#pragma unroll
  for (int s2 = 0; s2 < 2; s2++)
#pragma unroll
    for (int v = 0; v < 8; v++) creg[s2][v] = 0.f;
  __syncthreads();

  const half_t* W = Whh + (size_t)d * 512 * 128;
  int arow_l = lane & 15, akb = (lane >> 4) << 4;
  int bcol = lane & 15,  bkb = (lane >> 4) << 4;

  for (int n = 0; n < TT; n++) {
    int t = d ? (TT - 1 - n) : n;
    const half_t* hcur = hbuf[n & 1];
    half_t* hnxt = hbuf[(n & 1) ^ 1];

    for (int task = wv; task < 48; task += 32) {  // 6 mtiles x 8 jtiles
      int slot = task >> 5;
      int mt = task / 8, jt = task % 8;
      int rb = (mt << 4) + ((lane >> 4) << 3);
      int jj = (jt << 4) + (lane & 15);

      // warm L2 with next timestep's gate pre-activations (overlaps with WMMA chain)
      int tn = d ? t - 1 : t + 1;
      if (tn >= 0 && tn < TT) {
        size_t cb = (size_t)d * 512 + jj;
#pragma unroll
        for (int v = 0; v < 8; v++) {
          int s = rb + v;
          if (s < SEQ) {
            const float* gp = G + (size_t)(s * TT + tn) * 1024 + cb;
            __builtin_prefetch(gp, 0, 0);
            __builtin_prefetch(gp + 128, 0, 0);
            __builtin_prefetch(gp + 256, 0, 0);
            __builtin_prefetch(gp + 384, 0, 0);
          }
        }
      }

      v8f acc[4] = {};
      const half_t* Abase = hcur + (size_t)((mt << 4) + arow_l) * 128 + akb;
      const half_t* Wj = W + (size_t)((jt << 4) + bcol) * 128 + bkb;
#pragma unroll
      for (int kt = 0; kt < 4; kt++) {
        v16h a = ld_frag_a(Abase + kt * 32);
        v16h bf[4];
#pragma unroll
        for (int g = 0; g < 4; g++)                       // batch loads: 1 wait per kt
          bf[g] = ld_frag_b(Wj + ((size_t)(g << 7)) * 128 + kt * 32);
#pragma unroll
        for (int g = 0; g < 4; g++)
          acc[g] = __builtin_amdgcn_wmma_f32_16x16x32_f16(false, a, false, bf[g],
                                                          (short)0, acc[g], false, false);
      }

#pragma unroll
      for (int v = 0; v < 8; v++) {
        int s = rb + v;
        float gi = acc[0][v], gf = acc[1][v], gg = acc[2][v], go = acc[3][v];
        if (s < SEQ) {
          size_t gbase = (size_t)(s * TT + t) * 1024 + (size_t)d * 512 + jj;
          gi += G[gbase];  gf += G[gbase + 128];
          gg += G[gbase + 256]; go += G[gbase + 384];
        }
        float ii = 1.f / (1.f + __expf(-gi));
        float ff = 1.f / (1.f + __expf(-gf));
        float oo = 1.f / (1.f + __expf(-go));
        float cc = ff * creg[slot][v] + ii * tanhf(gg);
        creg[slot][v] = cc;
        float hh = oo * tanhf(cc);
        hnxt[(size_t)s * 128 + jj] = (half_t)hh;
        if (s < SEQ)
          Y[(size_t)(s * TT + t) * 256 + (size_t)d * 128 + jj] = hh;
      }
    }
    __syncthreads();
  }
}

// ---------- neighborhood attention: one wave per (query, head), online softmax ----------
__global__ __launch_bounds__(256)
void na2d_kernel(const float* __restrict__ qkv,   // (P_TOT, 768)
                 const float* __restrict__ rpb,   // (4, 9, 9)
                 half_t* __restrict__ attn16) {   // (P_TOT, 256)
  int w = (blockIdx.x * blockDim.x + threadIdx.x) >> 5;
  int lane = threadIdx.x & 31;
  if (w >= TT * SEQ * 4) return;
  int head = w & 3;
  int r = w >> 2;
  int i = r / SEQ, j = r % SEQ;
  int ihs = min(max(i - 2, 0), TT - 5);
  int iws = min(max(j - 2, 0), SEQ - 5);
  int d0 = lane << 1;
  const float* qp = qkv + (size_t)r * 768 + head * 64 + d0;
  float q0 = qp[0] * 0.125f, q1 = qp[1] * 0.125f;   // hd^-0.5 = 0.125
  float m = -1e30f, ssum = 0.f, a0 = 0.f, a1 = 0.f;
  for (int a = 0; a < 5; a++) {
    for (int b = 0; b < 5; b++) {
      int in_ = ihs + a, jn = iws + b;
      size_t rn = (size_t)(in_ * SEQ + jn) * 768 + head * 64 + d0;
      float p = q0 * qkv[rn + 256] + q1 * qkv[rn + 257];
#pragma unroll
      for (int off = 16; off > 0; off >>= 1) p += __shfl_xor(p, off, 32);
      float lg = p + rpb[head * 81 + (in_ - i + 4) * 9 + (jn - j + 4)];
      float mn = fmaxf(m, lg);
      float corr = __expf(m - mn);
      float wgt  = __expf(lg - mn);
      ssum = ssum * corr + wgt;
      a0 = a0 * corr + wgt * qkv[rn + 512];
      a1 = a1 * corr + wgt * qkv[rn + 513];
      m = mn;
    }
  }
  float inv = 1.f / ssum;
  size_t ob = (size_t)r * 256 + head * 64 + d0;
  attn16[ob]     = (half_t)(a0 * inv);
  attn16[ob + 1] = (half_t)(a1 * inv);
}

// ---------- elementwise / prep kernels ----------
__global__ void build_x0(const int* __restrict__ label, const float* __restrict__ feature,
                         const float* __restrict__ label_emb, half_t* __restrict__ X0h) {
  int idx = blockIdx.x * blockDim.x + threadIdx.x;
  if (idx >= P_TOT * CDIM) return;
  int k = idx & 255, rrow = idx >> 8;          // rrow = s*128 + t
  int s = rrow >> 7, t = rrow & 127;
  int p = t * SEQ + s;
  float val = (k < 128) ? label_emb[label[p] * 128 + k]
                        : feature[(size_t)(k - 128) * P_TOT + p];
  X0h[idx] = (half_t)val;
}

__global__ void f32_to_f16(const float* __restrict__ in, half_t* __restrict__ out, int n) {
  int idx = blockIdx.x * blockDim.x + threadIdx.x;
  if (idx < n) out[idx] = (half_t)in[idx];
}

// out[m][n][k] = in[m][k][n]  (in: count x rows(K) x cols(N))
__global__ void transpose_to_f16(const float* __restrict__ in, half_t* __restrict__ out,
                                 int rows, int cols, int count) {
  int idx = blockIdx.x * blockDim.x + threadIdx.x;
  int total = count * rows * cols;
  if (idx >= total) return;
  int m = idx / (rows * cols);
  int rem = idx % (rows * cols);
  int n = rem / rows, k = rem % rows;
  out[idx] = (half_t)in[(size_t)m * rows * cols + (size_t)k * cols + n];
}

__global__ void permute_y_to_x(const float* __restrict__ Y1, float* __restrict__ x) {
  int idx = blockIdx.x * blockDim.x + threadIdx.x;
  if (idx >= P_TOT * CDIM) return;
  int c = idx & 255, r2 = idx >> 8;            // r2 = t*88 + s
  int t = r2 / SEQ, s = r2 % SEQ;
  x[idx] = Y1[(size_t)(s * TT + t) * 256 + c];
}

__global__ void temb_add(const float* __restrict__ x, const float* __restrict__ temb,
                         const int* __restrict__ t_in, half_t* __restrict__ h16, int layer) {
  int idx = blockIdx.x * blockDim.x + threadIdx.x;
  if (idx >= P_TOT * CDIM) return;
  int c = idx & 255;
  int t0 = t_in[0];
  h16[idx] = (half_t)(x[idx] + temb[(size_t)layer * 1000 * 256 + (size_t)t0 * 256 + c]);
}

__global__ void out_proj(const float* __restrict__ x, const float* __restrict__ out_w,
                         const float* __restrict__ out_b, float* __restrict__ out) {
  int idx = blockIdx.x * blockDim.x + threadIdx.x;
  if (idx >= 5 * P_TOT) return;
  int r = idx % P_TOT, c = idx / P_TOT;
  float s = out_b[c];
  for (int k = 0; k < 256; k++) s += x[(size_t)r * 256 + k] * out_w[k * 5 + c];
  out[(size_t)c * P_TOT + r] = s;
}

// ---------- host launcher ----------
extern "C" void kernel_launch(void* const* d_in, const int* in_sizes, int n_in,
                              void* d_out, int out_size, void* d_ws, size_t ws_size,
                              hipStream_t stream) {
  const int*   label     = (const int*)d_in[0];
  const float* feature   = (const float*)d_in[1];
  const int*   t_in      = (const int*)d_in[2];
  const float* label_emb = (const float*)d_in[3];
  const float* lstm_Wih  = (const float*)d_in[4];
  const float* lstm_Whh  = (const float*)d_in[5];
  const float* lstm_b    = (const float*)d_in[6];
  const float* temb      = (const float*)d_in[7];
  const float* qkv_w     = (const float*)d_in[8];
  const float* qkv_b     = (const float*)d_in[9];
  const float* rpb       = (const float*)d_in[10];
  const float* proj_w    = (const float*)d_in[11];
  const float* proj_b    = (const float*)d_in[12];
  const float* out_w     = (const float*)d_in[13];
  const float* out_b     = (const float*)d_in[14];
  float* out = (float*)d_out;

  char* ws = (char*)d_ws;
  size_t off = 0;
  auto alloc = [&](size_t bytes) -> char* {
    char* p = ws + off;
    off += (bytes + 255) & ~(size_t)255;
    return p;
  };
  const size_t PC = (size_t)P_TOT * CDIM;
  half_t* X0h   = (half_t*)alloc(PC * 2);
  half_t* X1h   = (half_t*)alloc(PC * 2);
  half_t* Wih16 = (half_t*)alloc((size_t)2 * 2 * 512 * 256 * 2);
  half_t* Whh16 = (half_t*)alloc((size_t)2 * 2 * 512 * 128 * 2);
  half_t* qkvT  = (half_t*)alloc((size_t)2 * 768 * 256 * 2);
  half_t* projT = (half_t*)alloc((size_t)2 * 256 * 256 * 2);
  float*  G     = (float*)alloc((size_t)P_TOT * 1024 * 4);
  float*  Y0    = (float*)alloc(PC * 4);
  float*  Y1    = (float*)alloc(PC * 4);
  float*  xbuf  = (float*)alloc(PC * 4);
  // aliases with disjoint lifetimes:
  float*  qkvbuf = G;      // G dead after lstm scans; qkv lives only in attention
  half_t* h16    = X1h;    // X1h dead after layer-1 input GEMM
  half_t* attn16 = X0h;    // X0h dead after layer-0 input GEMM

  const int TB = 256;
  int nElem = P_TOT * CDIM;
  int ecBlocks = (nElem + TB - 1) / TB;

  // prep
  build_x0<<<ecBlocks, TB, 0, stream>>>(label, feature, label_emb, X0h);
  f32_to_f16<<<(2 * 2 * 512 * 256 + TB - 1) / TB, TB, 0, stream>>>(lstm_Wih, Wih16, 2 * 2 * 512 * 256);
  f32_to_f16<<<(2 * 2 * 512 * 128 + TB - 1) / TB, TB, 0, stream>>>(lstm_Whh, Whh16, 2 * 2 * 512 * 128);
  transpose_to_f16<<<(2 * 768 * 256 + TB - 1) / TB, TB, 0, stream>>>(qkv_w, qkvT, 256, 768, 2);
  transpose_to_f16<<<(2 * 256 * 256 + TB - 1) / TB, TB, 0, stream>>>(proj_w, projT, 256, 256, 2);

  auto launch_gemm = [&](const half_t* A, const half_t* Bt, const float* bias,
                         const float* res, float* C, half_t* Ch, int M, int N, int K) {
    int total = (M / 16) * (N / 64);
    int blocks = (total + 7) / 8;   // 8 waves per 256-thread block
    gemm_wmma<<<blocks, TB, 0, stream>>>(A, Bt, bias, res, C, Ch, M, N, K);
  };

  // LSTM layer 0: G = X0 @ Wih0^T + b0 (both dirs fused, N=1024), then scan
  launch_gemm(X0h, Wih16, lstm_b, nullptr, G, nullptr, P_TOT, 1024, 256);
  lstm_scan<<<2, 1024, 0, stream>>>(G, Whh16, Y0);
  f32_to_f16<<<ecBlocks, TB, 0, stream>>>(Y0, X1h, nElem);

  // LSTM layer 1
  launch_gemm(X1h, Wih16 + 2 * 512 * 256, lstm_b + 1024, nullptr, G, nullptr, P_TOT, 1024, 256);
  lstm_scan<<<2, 1024, 0, stream>>>(G, Whh16 + 2 * 512 * 128, Y1);

  // (s,t,c) -> (t,s,c) residual stream
  permute_y_to_x<<<ecBlocks, TB, 0, stream>>>(Y1, xbuf);

  // attention blocks
  for (int l = 0; l < 2; l++) {
    temb_add<<<ecBlocks, TB, 0, stream>>>(xbuf, temb, t_in, h16, l);
    launch_gemm(h16, qkvT + (size_t)l * 768 * 256, qkv_b + l * 768, nullptr,
                qkvbuf, nullptr, P_TOT, 768, 256);
    na2d_kernel<<<(TT * SEQ * 4) / 8, TB, 0, stream>>>(qkvbuf, rpb + l * 4 * 81, attn16);
    launch_gemm(attn16, projT + (size_t)l * 256 * 256, proj_b + l * 256, xbuf,
                xbuf, nullptr, P_TOT, 256, 256);
  }

  // final projection to 5 classes, output (1, 5, 11264)
  out_proj<<<(5 * P_TOT + TB - 1) / TB, TB, 0, stream>>>(xbuf, out_w, out_b, out);
}
